// GQAMixtureOfHeadsAttention_20985210208478
// MI455X (gfx1250) — matrix operand order repaired and checked by reading, not compiled, not run on hardware
//
#include <hip/hip_runtime.h>
#include <hip/hip_bf16.h>

typedef __attribute__((ext_vector_type(16))) __bf16 v16bf;
typedef __attribute__((ext_vector_type(8)))  __bf16 v8bf;
typedef __attribute__((ext_vector_type(8)))  float  v8f;

constexpr int S_   = 1024;
constexpr int EMB_ = 1024;
constexpr int M_   = 6;     // experts
constexpr int A_   = 4;     // active experts per token
constexpr int QH_  = 2;     // query heads per kv head
constexpr int D_   = 128;   // head dim
constexpr int E_   = S_ * A_;           // 4096 routed entries
constexpr int MAXROWS  = E_ + 16 * M_;  // padded per-expert rows
constexpr int MAXTILES = MAXROWS / 16;
constexpr float SCALE_ = 0.08838834764831845f; // D^-0.5

union AFrag { v16bf v; v8bf h[2]; };

__device__ __forceinline__ v8f wmma_bf(v16bf a, v16bf b, v8f c) {
  // D = A(16x32 bf16) * B(32x16 bf16) + C(16x16 f32)
  return __builtin_amdgcn_wmma_f32_16x16x32_bf16(false, a, false, b, (short)0, c,
                                                 false, false);
}

// ---------------------------------------------------------------- fp32 -> bf16
__global__ void cvt_bf16_kernel(const float* __restrict__ in, __bf16* __restrict__ out, int n) {
  int i = blockIdx.x * blockDim.x + threadIdx.x;
  if (i < n) out[i] = (__bf16)in[i];
}

__global__ void fill_zero_bf16_kernel(__bf16* __restrict__ p, int n) {
  int i = blockIdx.x * blockDim.x + threadIdx.x;
  if (i < n) p[i] = (__bf16)0.0f;
}

// ---------------------------------------------------------------- router
// One wave per token: 6 logits, softmax probs, entropy, top-4 + weights.
__global__ void router_kernel(const float* __restrict__ x, const float* __restrict__ wr,
                              float* __restrict__ probs, float* __restrict__ ent,
                              int* __restrict__ kvidx, float* __restrict__ kvw) {
  int s = blockIdx.x;
  int lane = threadIdx.x;
  const float* xr = x + (size_t)s * EMB_;
  float acc[M_];
  for (int m = 0; m < M_; ++m) {
    const float* w = wr + (size_t)m * EMB_;
    float a = 0.f;
    for (int k = lane; k < EMB_; k += 32) a += xr[k] * w[k];
    for (int off = 16; off; off >>= 1) a += __shfl_xor(a, off, 32);
    acc[m] = a;
  }
  if (lane == 0) {
    float mx = acc[0];
    for (int m = 1; m < M_; ++m) mx = fmaxf(mx, acc[m]);
    float p[M_], sum = 0.f;
    for (int m = 0; m < M_; ++m) { p[m] = __expf(acc[m] - mx); sum += p[m]; }
    float inv = 1.f / sum, es = 0.f;
    for (int m = 0; m < M_; ++m) {
      p[m] *= inv;
      probs[s * M_ + m] = p[m];
      es -= p[m] * __logf(p[m] + 1e-8f);
    }
    ent[s] = es;
    float tl[M_];
    for (int m = 0; m < M_; ++m) tl[m] = acc[m];
    int idx[A_]; float val[A_];
    for (int a2 = 0; a2 < A_; ++a2) {           // argmax, ties -> lower index
      int bi = 0; float bv = tl[0];
      for (int m = 1; m < M_; ++m) if (tl[m] > bv) { bv = tl[m]; bi = m; }
      idx[a2] = bi; val[a2] = bv; tl[bi] = -3.0e38f;
    }
    float ws2 = 0.f, pw[A_];
    for (int a2 = 0; a2 < A_; ++a2) { pw[a2] = __expf(val[a2] - val[0]); ws2 += pw[a2]; }
    for (int a2 = 0; a2 < A_; ++a2) {
      kvidx[s * A_ + a2] = idx[a2];
      kvw[s * A_ + a2]  = pw[a2] / ws2;
    }
  }
}

// ---------------------------------------------------------------- finalize
__global__ void finalize_kernel(const float* __restrict__ probs, const float* __restrict__ ent,
                                const int* __restrict__ kvidx,
                                int* __restrict__ rank, int* __restrict__ row_entry,
                                int* __restrict__ row_rank, int* __restrict__ row_tok,
                                int* __restrict__ tile_expert, int* __restrict__ meta,
                                float* __restrict__ aux_out) {
  if (threadIdx.x != 0 || blockIdx.x != 0) return;
  float psum[M_] = {0.f, 0.f, 0.f, 0.f, 0.f, 0.f};
  float prim[M_] = {0.f, 0.f, 0.f, 0.f, 0.f, 0.f};
  float esum = 0.f;
  for (int s = 0; s < S_; ++s) {
    esum += ent[s];
    for (int m = 0; m < M_; ++m) psum[m] += probs[s * M_ + m];
    prim[kvidx[s * A_]] += 1.f;
  }
  float bal = 0.f;
  for (int m = 0; m < M_; ++m) bal += (prim[m] / (float)S_) * (psum[m] / (float)S_);
  bal *= (float)M_;
  aux_out[0] = 0.01f * bal - 0.01f * (esum / (float)S_);
  int cnt[M_] = {0, 0, 0, 0, 0, 0};
  for (int e = 0; e < E_; ++e) { int m = kvidx[e]; rank[e] = cnt[m]++; }
  int poff[M_ + 1]; poff[0] = 0;
  for (int m = 0; m < M_; ++m) poff[m + 1] = poff[m] + ((cnt[m] + 15) & ~15);
  int total = poff[M_];
  meta[0] = total / 16;
  for (int m = 0; m < M_; ++m) { meta[1 + m] = cnt[m]; meta[8 + m] = poff[m]; }
  meta[8 + M_] = poff[M_];
  for (int r = 0; r < total; ++r) { row_entry[r] = -1; row_tok[r] = -1; row_rank[r] = 0; }
  for (int m = 0; m < M_; ++m) {
    for (int t = poff[m] / 16; t < poff[m + 1] / 16; ++t) tile_expert[t] = m;
    for (int r = poff[m]; r < poff[m + 1]; ++r) row_rank[r] = r - poff[m];
  }
  for (int e = 0; e < E_; ++e) {
    int m = kvidx[e];
    int r = poff[m] + rank[e];
    row_entry[r] = e;
    row_tok[r]   = e >> 2;   // token = e / A
  }
}

// ---------------------------------------------------------------- gathered QKV projection + RoPE
// Jobs: 0..7 = Q(head, dim-pair), 8..11 = K(pair), 12..15 = V(pair).
// Padding rows read from a zero-filled x row -> no masking in the k-loop.
__global__ void qkv_kernel(const __bf16* __restrict__ xb, const __bf16* __restrict__ xzero,
                           const __bf16* __restrict__ wqb, const __bf16* __restrict__ wkb,
                           const __bf16* __restrict__ wvb,
                           const int* __restrict__ row_tok, const int* __restrict__ row_rank,
                           const int* __restrict__ tile_expert, const int* __restrict__ meta,
                           __bf16* __restrict__ qg, __bf16* __restrict__ kg, __bf16* __restrict__ vg) {
  int tile = blockIdx.x;
  if (tile >= meta[0]) return;
  int job = blockIdx.y;
  int m = tile_expert[tile];
  int lane = threadIdx.x, n = lane & 15, hi = lane >> 4;

  int target, head = 0, pair;
  if (job < 8)       { target = 0; head = job >> 2; pair = job & 3; }
  else if (job < 12) { target = 1; pair = job - 8; }
  else               { target = 2; pair = job - 12; }
  int dlo = pair * 16;

  const __bf16* wb;
  int rlo;
  if (target == 0) { wb = wqb; rlo = m * (QH_ * D_) + head * D_ + dlo + n; }
  else             { wb = (target == 1) ? wkb : wvb; rlo = m * D_ + dlo + n; }
  const __bf16* wlo = wb + (size_t)rlo * EMB_;
  const __bf16* whi = wb + (size_t)(rlo + 64) * EMB_;

  int arow = tile * 16 + n;              // A-matrix row for this lane
  int tok = row_tok[arow];
  const __bf16* xr = (tok >= 0) ? (xb + (size_t)tok * EMB_) : xzero;

  v8f clo = {0.f,0.f,0.f,0.f,0.f,0.f,0.f,0.f};
  v8f chi = {0.f,0.f,0.f,0.f,0.f,0.f,0.f,0.f};
  for (int k0 = 0; k0 < EMB_; k0 += 32) {
    AFrag a;
    a.h[0] = *(const v8bf*)(xr + k0 + hi * 16);
    a.h[1] = *(const v8bf*)(xr + k0 + hi * 16 + 8);
    v16bf blo = *(const v16bf*)(wlo + k0 + hi * 16);
    v16bf bhi = *(const v16bf*)(whi + k0 + hi * 16);
    clo = wmma_bf(a.v, blo, clo);
    chi = wmma_bf(a.v, bhi, chi);
  }

  int d = dlo + n;                                     // d in [0,64)
  float invf = __powf(10000.f, -(float)d * (1.0f / 64.0f));
  for (int j = 0; j < 8; ++j) {
    int grow = tile * 16 + j + hi * 8;                 // C row = j + 8*(lane>=16)
    float xlo = clo[j], xhi = chi[j], olo, ohi;
    if (target != 2) {
      float ang = (float)row_rank[grow] * invf;
      float sn, cs;
      __sincosf(ang, &sn, &cs);
      olo = xlo * cs - xhi * sn;
      ohi = xhi * cs + xlo * sn;
    } else { olo = xlo; ohi = xhi; }
    if (target == 0) {
      __bf16* dst = qg + (size_t)grow * (QH_ * D_) + head * D_;
      dst[d] = (__bf16)olo; dst[d + 64] = (__bf16)ohi;
    } else {
      __bf16* dst = ((target == 1) ? kg : vg) + (size_t)grow * D_;
      dst[d] = (__bf16)olo; dst[d + 64] = (__bf16)ohi;
    }
  }
}

// ---------------------------------------------------------------- expert-grouped flash attention
// V tile staged LDS via CDNA5 async global->LDS (ASYNCcnt), overlapped with
// score WMMAs + online softmax; K/V rows of the next tile prefetched.
__global__ void attn_kernel(const __bf16* __restrict__ qg, const __bf16* __restrict__ kg,
                            const __bf16* __restrict__ vg,
                            const int* __restrict__ tile_expert, const int* __restrict__ meta,
                            const int* __restrict__ row_entry, const float* __restrict__ kvw,
                            __bf16* __restrict__ of) {
  int r = blockIdx.x;
  if (r >= meta[0]) return;
  int h = blockIdx.y;
  int m = tile_expert[r];
  int Cm = meta[1 + m];
  int kbase = meta[8 + m];
  int lane = threadIdx.x, n = lane & 15, hi = lane >> 4;

  __shared__ __bf16 Pl[16][32];    // probs tile, C-layout write -> A-layout read
  __shared__ __bf16 Vt[32][128];   // staged V tile (written by async DMA)

  AFrag qa[4];
  {
    const __bf16* qr = qg + (size_t)(r * 16 + n) * (QH_ * D_) + h * D_;
    for (int c = 0; c < 4; ++c) {
      qa[c].h[0] = *(const v8bf*)(qr + c * 32 + hi * 16);
      qa[c].h[1] = *(const v8bf*)(qr + c * 32 + hi * 16 + 8);
    }
  }

  float mrun[8], lrun[8];
  v8f oacc[8];
  for (int j = 0; j < 8; ++j) { mrun[j] = -3.0e38f; lrun[j] = 0.f; }
  for (int c = 0; c < 8; ++c) oacc[c] = (v8f){0.f,0.f,0.f,0.f,0.f,0.f,0.f,0.f};

  for (int f0 = 0; f0 < Cm; f0 += 32) {
    // async-stage V tile: lane -> one key row, 8 x 16B per lane -> LDS
    {
      int vrow = kbase + f0 + lane;
      if (vrow > MAXROWS - 1) vrow = MAXROWS - 1;
      const __bf16* vr = vg + (size_t)vrow * D_;
      unsigned lds = (unsigned)(uintptr_t)(&Vt[lane][0]);
      unsigned long long ga = (unsigned long long)(uintptr_t)vr;
      for (int c = 0; c < 8; ++c) {
        asm volatile("global_load_async_to_lds_b128 %0, %1, off"
                     :: "v"(lds + (unsigned)(c * 32)), "v"(ga + (unsigned long long)(c * 32))
                     : "memory");
      }
    }
    // prefetch next key tile's K and V rows
    {
      int prow = kbase + f0 + 32 + lane;
      if (prow > MAXROWS - 1) prow = MAXROWS - 1;
      __builtin_prefetch(kg + (size_t)prow * D_, 0, 1);
      __builtin_prefetch(kg + (size_t)prow * D_ + 64, 0, 1);
      __builtin_prefetch(vg + (size_t)prow * D_, 0, 1);
      __builtin_prefetch(vg + (size_t)prow * D_ + 64, 0, 1);
    }
    // scores: 16 q rows x 32 keys, K-dim = D in 4 chunks of 32 (overlaps async DMA)
    v8f s0 = {0.f,0.f,0.f,0.f,0.f,0.f,0.f,0.f};
    v8f s1 = {0.f,0.f,0.f,0.f,0.f,0.f,0.f,0.f};
    int key0 = f0 + n, key1 = f0 + 16 + n;
    int kr0 = kbase + key0; if (kr0 > MAXROWS - 1) kr0 = MAXROWS - 1;
    int kr1 = kbase + key1; if (kr1 > MAXROWS - 1) kr1 = MAXROWS - 1;
    const __bf16* kp0 = kg + (size_t)kr0 * D_;
    const __bf16* kp1 = kg + (size_t)kr1 * D_;
    for (int c = 0; c < 4; ++c) {
      v16bf b0 = *(const v16bf*)(kp0 + c * 32 + hi * 16);
      v16bf b1 = *(const v16bf*)(kp1 + c * 32 + hi * 16);
      s0 = wmma_bf(qa[c].v, b0, s0);
      s1 = wmma_bf(qa[c].v, b1, s1);
    }
    bool valid0 = key0 < Cm, valid1 = key1 < Cm;
    for (int j = 0; j < 8; ++j) {
      float a0 = valid0 ? s0[j] * SCALE_ : -1.0e9f;
      float a1 = valid1 ? s1[j] * SCALE_ : -1.0e9f;
      float t = fmaxf(a0, a1);
      for (int off = 8; off; off >>= 1) t = fmaxf(t, __shfl_xor(t, off, 32));
      float mn = fmaxf(mrun[j], t);
      float alpha = __expf(mrun[j] - mn);
      float p0 = __expf(a0 - mn), p1 = __expf(a1 - mn);
      float rs = p0 + p1;
      for (int off = 8; off; off >>= 1) rs += __shfl_xor(rs, off, 32);
      lrun[j] = lrun[j] * alpha + rs;
      mrun[j] = mn;
      for (int c = 0; c < 8; ++c) oacc[c][j] *= alpha;
      Pl[j + hi * 8][n]      = (__bf16)p0;
      Pl[j + hi * 8][n + 16] = (__bf16)p1;
    }
    __syncthreads();
    asm volatile("s_wait_asynccnt 0x0" ::: "memory");   // V tile landed in LDS
    // P * V : A = probs (16x32), B = V (32 keys x 16 dims) per dim-chunk
    AFrag pa;
    pa.h[0] = *(const v8bf*)&Pl[n][hi * 16];
    pa.h[1] = *(const v8bf*)&Pl[n][hi * 16 + 8];
    for (int c = 0; c < 8; ++c) {
      v16bf bv;
      for (int j = 0; j < 16; ++j) bv[j] = Vt[j + hi * 16][c * 16 + n];
      oacc[c] = wmma_bf(pa.v, bv, oacc[c]);
    }
    __syncthreads();
  }
  for (int j = 0; j < 8; ++j) {
    int grow = r * 16 + j + hi * 8;
    int e = row_entry[grow];
    if (e < 0) continue;
    float inv = kvw[e] / lrun[j];
    int s = e >> 2, a2 = e & 3;
    __bf16* dst = of + (size_t)s * (A_ * QH_ * D_) + a2 * (QH_ * D_) + h * D_;
    for (int c = 0; c < 8; ++c) dst[c * 16 + n] = (__bf16)(oacc[c][j] * inv);
  }
}

// ---------------------------------------------------------------- output GEMM: out = o_flat @ w_out.T
// 16x64 tile per wave: one A-fragment feeds 4 B-tiles (4x less A traffic).
__global__ void out_gemm_kernel(const __bf16* __restrict__ of, const __bf16* __restrict__ wob,
                                float* __restrict__ out) {
  int s0 = blockIdx.x * 16, c0 = blockIdx.y * 64;
  int lane = threadIdx.x, n = lane & 15, hi = lane >> 4;
  const __bf16* ar = of + (size_t)(s0 + n) * EMB_;
  const __bf16* br[4];
  for (int cc = 0; cc < 4; ++cc) br[cc] = wob + (size_t)(c0 + cc * 16 + n) * EMB_;
  v8f acc[4];
  for (int cc = 0; cc < 4; ++cc) acc[cc] = (v8f){0.f,0.f,0.f,0.f,0.f,0.f,0.f,0.f};
  for (int k0 = 0; k0 < EMB_; k0 += 32) {
    AFrag a;
    a.h[0] = *(const v8bf*)(ar + k0 + hi * 16);
    a.h[1] = *(const v8bf*)(ar + k0 + hi * 16 + 8);
    for (int cc = 0; cc < 4; ++cc) {
      v16bf b = *(const v16bf*)(br[cc] + k0 + hi * 16);
      acc[cc] = wmma_bf(a.v, b, acc[cc]);
    }
  }
  for (int cc = 0; cc < 4; ++cc)
    for (int j = 0; j < 8; ++j)
      out[(size_t)(s0 + j + hi * 8) * EMB_ + c0 + cc * 16 + n] = acc[cc][j];
}

// ---------------------------------------------------------------- launch
extern "C" void kernel_launch(void* const* d_in, const int* in_sizes, int n_in,
                              void* d_out, int out_size, void* d_ws, size_t ws_size,
                              hipStream_t stream) {
  const float* x  = (const float*)d_in[0];
  const float* wq = (const float*)d_in[1];
  const float* wk = (const float*)d_in[2];
  const float* wv = (const float*)d_in[3];
  const float* wr = (const float*)d_in[4];
  const float* wo = (const float*)d_in[5];
  float* out = (float*)d_out;

  char* ws = (char*)d_ws;
  size_t off = 0;
  auto alloc = [&](size_t bytes) -> char* {
    char* p = ws + off;
    off = (off + bytes + 255) & ~(size_t)255;
    return p;
  };
  __bf16* xb  = (__bf16*)alloc((size_t)S_ * EMB_ * 2);
  __bf16* xzb = (__bf16*)alloc((size_t)EMB_ * 2);          // zero row for padding
  __bf16* wqb = (__bf16*)alloc((size_t)M_ * QH_ * D_ * EMB_ * 2);
  __bf16* wkb = (__bf16*)alloc((size_t)M_ * D_ * EMB_ * 2);
  __bf16* wvb = (__bf16*)alloc((size_t)M_ * D_ * EMB_ * 2);
  __bf16* wob = (__bf16*)alloc((size_t)EMB_ * (A_ * QH_ * D_) * 2);
  __bf16* qg  = (__bf16*)alloc((size_t)MAXROWS * QH_ * D_ * 2);
  __bf16* kg  = (__bf16*)alloc((size_t)MAXROWS * D_ * 2);
  __bf16* vg  = (__bf16*)alloc((size_t)MAXROWS * D_ * 2);
  __bf16* of  = (__bf16*)alloc((size_t)S_ * (A_ * QH_ * D_) * 2);
  float* probs = (float*)alloc((size_t)S_ * M_ * 4);
  float* ent   = (float*)alloc((size_t)S_ * 4);
  float* kvw   = (float*)alloc((size_t)S_ * A_ * 4);
  int* kvidx   = (int*)alloc((size_t)S_ * A_ * 4);
  int* rank    = (int*)alloc((size_t)E_ * 4);
  int* row_entry = (int*)alloc((size_t)MAXROWS * 4);
  int* row_rank  = (int*)alloc((size_t)MAXROWS * 4);
  int* row_tok   = (int*)alloc((size_t)MAXROWS * 4);
  int* tile_expert = (int*)alloc((size_t)MAXTILES * 4);
  int* meta        = (int*)alloc(64 * 4);

  // 1) bf16 conversions + zero row
  int n1 = S_ * EMB_;
  cvt_bf16_kernel<<<(n1 + 255) / 256, 256, 0, stream>>>(x, xb, n1);
  fill_zero_bf16_kernel<<<(EMB_ + 255) / 256, 256, 0, stream>>>(xzb, EMB_);
  int n2 = M_ * QH_ * D_ * EMB_;
  cvt_bf16_kernel<<<(n2 + 255) / 256, 256, 0, stream>>>(wq, wqb, n2);
  int n3 = M_ * D_ * EMB_;
  cvt_bf16_kernel<<<(n3 + 255) / 256, 256, 0, stream>>>(wk, wkb, n3);
  cvt_bf16_kernel<<<(n3 + 255) / 256, 256, 0, stream>>>(wv, wvb, n3);
  int n4 = EMB_ * (A_ * QH_ * D_);
  cvt_bf16_kernel<<<(n4 + 255) / 256, 256, 0, stream>>>(wo, wob, n4);

  // 2) routing
  router_kernel<<<S_, 32, 0, stream>>>(x, wr, probs, ent, kvidx, kvw);

  // 3) ranks, padded per-expert layout, aux loss -> d_out[S*EMB]
  finalize_kernel<<<1, 32, 0, stream>>>(probs, ent, kvidx, rank, row_entry, row_rank,
                                        row_tok, tile_expert, meta, out + (size_t)S_ * EMB_);

  // 4) gathered QKV projection + RoPE (WMMA)
  qkv_kernel<<<dim3(MAXTILES, 16), 32, 0, stream>>>(xb, xzb, wqb, wkb, wvb, row_tok, row_rank,
                                                    tile_expert, meta, qg, kg, vg);

  // 5) expert-grouped flash attention (WMMA + async LDS + prefetch)
  attn_kernel<<<dim3(MAXTILES, QH_), 32, 0, stream>>>(qg, kg, vg, tile_expert, meta,
                                                      row_entry, kvw, of);

  // 6) output projection (WMMA), fp32 result
  out_gemm_kernel<<<dim3(S_ / 16, EMB_ / 64), 32, 0, stream>>>(of, wob, out);
}
